// EfficientVitBlock_45930380263473
// MI455X (gfx1250) — matrix operand order repaired and checked by reading, not compiled, not run on hardware
//
#include <hip/hip_runtime.h>
#include <stdint.h>

// ---------- types ----------
typedef __bf16 bf16_t;
typedef bf16_t  v16bf __attribute__((ext_vector_type(16)));
typedef float   v8f   __attribute__((ext_vector_type(8)));
typedef uint32_t v4u  __attribute__((ext_vector_type(4)));
typedef int      v8i  __attribute__((ext_vector_type(8)));
typedef int      v4i  __attribute__((ext_vector_type(4)));

#define WMMA_BF16(A, B, C) \
  __builtin_amdgcn_wmma_f32_16x16x32_bf16(false, (A), false, (B), (short)0, (C), false, false)

// ---------- TDM: tensor_load_to_lds with a hand-built 2D descriptor ----------
__device__ __forceinline__ uint32_t lds_addr_of(const void* p) {
  // LDS aperture base has zero low-32 bits; flat-addr[31:0] == LDS byte offset.
  return (uint32_t)(uintptr_t)p;
}

__device__ __forceinline__ void tdm_load_2d(uint32_t lds, const float* gptr,
                                            uint32_t tile0, uint32_t tile1,
                                            uint64_t stride0_elems) {
  uint64_t ga = (uint64_t)(uintptr_t)gptr;
  const uint32_t td0 = 0x7FFFFFFFu, td1 = 0x7FFFFFFFu;  // no OOB clipping needed
  v4u g0;
  g0[0] = 1u;                                   // count=1, user descriptor
  g0[1] = lds;                                  // lds_addr (bytes)
  g0[2] = (uint32_t)ga;                         // global_addr[31:0]
  g0[3] = (uint32_t)((ga >> 32) & 0x01FFFFFFu)  // global_addr[56:32]
        | (2u << 30);                           // type = 2 ("image")
  v8i g1;
  g1[0] = (int)(2u << 16);                      // data_size = 2 -> 4 bytes
  g1[1] = (int)((td0 & 0xFFFFu) << 16);         // tensor_dim0[15:0]
  g1[2] = (int)((td0 >> 16) | ((td1 & 0xFFFFu) << 16));
  g1[3] = (int)((td1 >> 16) | (tile0 << 16));   // tile_dim0
  g1[4] = (int)(tile1 & 0xFFFFu);               // tile_dim1 (tile_dim2 = 0)
  g1[5] = (int)(uint32_t)(stride0_elems & 0xFFFFFFFFull);   // dim0 stride lo
  g1[6] = (int)(uint32_t)((stride0_elems >> 32) & 0xFFFFull);
  g1[7] = 0;
  v4i z4 = {0, 0, 0, 0};
#if defined(__clang_major__) && (__clang_major__ >= 23)
  v8i z8 = {0, 0, 0, 0, 0, 0, 0, 0};
  __builtin_amdgcn_tensor_load_to_lds(g0, g1, z4, z4, z8, 0);
#else
  __builtin_amdgcn_tensor_load_to_lds(g0, g1, z4, z4, 0);
#endif
}

// ---------- WMMA fragment gathers (fp32 in LDS -> bf16 frags) ----------
// A (16x32): lane L<16 holds row L (K 0-15), L>=16 holds row L-16 (K 16-31)
__device__ __forceinline__ v16bf load_a_frag(const float* t, int row0, int ldk) {
  const int lane = threadIdx.x & 31;
  const int g = lane >> 4;
  const float* r = t + (size_t)(row0 + (lane & 15)) * ldk + g * 16;
  v16bf a;
#pragma unroll
  for (int h = 0; h < 8; ++h) a[h] = (bf16_t)r[h];
#pragma unroll
  for (int h = 0; h < 8; ++h) a[h + 8] = (bf16_t)r[8 + h];
  return a;
}
// B (32x16) from LDS laid [K rows][ldn cols]: lane holds col lane&15, K = 16*(lane>>4)+h
__device__ __forceinline__ v16bf load_b_frag(const float* t, int col0, int ldn) {
  const int lane = threadIdx.x & 31;
  const float* c = t + (size_t)((lane >> 4) * 16) * ldn + col0 + (lane & 15);
  v16bf b;
#pragma unroll
  for (int h = 0; h < 16; ++h) b[h] = (bf16_t)c[(size_t)h * ldn];
  return b;
}
// B (32x16) from LDS stored transposed [N rows][ldk cols of K]
__device__ __forceinline__ v16bf load_b_fragT(const float* t, int col0, int ldk) {
  const int lane = threadIdx.x & 31;
  const float* c = t + (size_t)(col0 + (lane & 15)) * ldk + (lane >> 4) * 16;
  v16bf b;
#pragma unroll
  for (int h = 0; h < 16; ++h) b[h] = (bf16_t)c[h];
  return b;
}

__device__ __forceinline__ float hswish(float v) {
  float u = v + 3.0f;
  u = u < 0.0f ? 0.0f : (u > 6.0f ? 6.0f : u);
  return v * u * (1.0f / 6.0f);
}

// ---------- Generic 1x1-conv GEMM: O[b,o,p] = sum_i W[o,i] X[b,i,p]  (+epilogue) ----------
// Block tile: M=64 (gridDim.y tiles) x N=128 (8 tiles/batch), K-steps of 32, TDM double-buffer.
// EPI: 0 = none, 1 = +bias, hardswish, 2 = *scale + bias + residual
template <int EPI>
__global__ __launch_bounds__(256) void gemm_wmma_kernel(
    const float* __restrict__ W, const float* __restrict__ X, float* __restrict__ O,
    const float* __restrict__ P0, const float* __restrict__ P1,
    const float* __restrict__ R, int K, int M) {
  __shared__ float sA[2][64 * 32];
  __shared__ float sB[2][32 * 128];
  const int wave = threadIdx.x >> 5, lane = threadIdx.x & 31;
  const int b = blockIdx.x >> 3, nt = blockIdx.x & 7;
  const int mt = blockIdx.y;
  const int wm = wave >> 1, wn = wave & 1;
  const float* Wt0 = W + (size_t)(mt * 64) * K;                 // 64 rows, stride K
  const float* Xt0 = X + ((size_t)b * K) * 1024 + nt * 128;     // 32 rows/step, stride 1024

  v8f acc[4];
#pragma unroll
  for (int t = 0; t < 4; ++t) {
    v8f z = {0.f, 0.f, 0.f, 0.f, 0.f, 0.f, 0.f, 0.f};
    acc[t] = z;
  }

  if (wave == 0) {
    tdm_load_2d(lds_addr_of(&sA[0][0]), Wt0, 32u, 64u, (uint64_t)K);
    tdm_load_2d(lds_addr_of(&sB[0][0]), Xt0, 128u, 32u, 1024ull);
    __builtin_amdgcn_s_wait_tensorcnt(0);
  }
  __syncthreads();

  const int steps = K >> 5;
  for (int s = 0; s < steps; ++s) {
    const int cur = s & 1, nxt = cur ^ 1;
    if (wave == 0 && (s + 1) < steps) {
      const int k0 = (s + 1) << 5;
      tdm_load_2d(lds_addr_of(&sA[nxt][0]), Wt0 + k0, 32u, 64u, (uint64_t)K);
      tdm_load_2d(lds_addr_of(&sB[nxt][0]), Xt0 + (size_t)k0 * 1024, 128u, 32u, 1024ull);
    }
    v16bf a = load_a_frag(&sA[cur][0], wm * 16, 32);
#pragma unroll
    for (int t = 0; t < 4; ++t) {
      v16bf bb = load_b_frag(&sB[cur][0], wn * 64 + t * 16, 128);
      acc[t] = WMMA_BF16(a, bb, acc[t]);
    }
    __syncthreads();                                   // everyone done reading 'cur'
    if (wave == 0 && (s + 1) < steps) __builtin_amdgcn_s_wait_tensorcnt(0);
    __syncthreads();                                   // 'nxt' buffer ready
  }

#pragma unroll
  for (int t = 0; t < 4; ++t) {
    const int pg = nt * 128 + wn * 64 + t * 16 + (lane & 15);
#pragma unroll
    for (int r = 0; r < 8; ++r) {
      const int o = mt * 64 + wm * 16 + (lane >> 4) * 8 + r;
      float v = acc[t][r];
      const size_t idx = ((size_t)b * M + o) * 1024 + pg;
      if constexpr (EPI == 1) {
        v = hswish(v + P0[o]);
      } else if constexpr (EPI == 2) {
        v = v * P0[o] + P1[o] + R[idx];
      }
      O[idx] = v;
    }
  }
}

// ---------- agg: depthwise 5x5 (VALU) fused with grouped 1x1 (WMMA, 24 groups of 32->32) ----------
__global__ __launch_bounds__(256) void agg_kernel(
    const float* __restrict__ QKV, const float* __restrict__ Wdw,
    const float* __restrict__ Wpw, float* __restrict__ AGG) {
  __shared__ float sdw[32 * 128];
  __shared__ float swt[32 * 32];
  const int tid = threadIdx.x, wave = tid >> 5, lane = tid & 31;
  const int b = blockIdx.x / 24, g = blockIdx.x % 24;
  const int pt = blockIdx.y;  // 128-pixel tile (4 image rows)

  for (int i = tid; i < 32 * 32; i += 256) swt[i] = Wpw[(size_t)(g * 32) * 32 + i];

  for (int i = tid; i < 32 * 128; i += 256) {
    const int cl = i >> 7, pl = i & 127;
    const int c = g * 32 + cl;
    const int px = pt * 128 + pl;
    const int y = px >> 5, x = px & 31;
    const float* im = QKV + ((size_t)b * 768 + c) * 1024;
    const float* wd = Wdw + c * 25;
    float s = 0.f;
#pragma unroll
    for (int dy = -2; dy <= 2; ++dy) {
      const int yy = y + dy;
      if (yy < 0 || yy > 31) continue;
#pragma unroll
      for (int dx = -2; dx <= 2; ++dx) {
        const int xx = x + dx;
        if (xx < 0 || xx > 31) continue;
        s += im[yy * 32 + xx] * wd[(dy + 2) * 5 + (dx + 2)];
      }
    }
    sdw[i] = s;
  }
  __syncthreads();

  for (int job = wave; job < 16; job += 8) {     // 2 M-tiles x 8 N-tiles
    const int mt2 = job >> 3, nt2 = job & 7;
    v16bf a = load_a_frag(swt, mt2 * 16, 32);
    v16bf bb = load_b_frag(sdw, nt2 * 16, 128);
    v8f c0 = {0.f, 0.f, 0.f, 0.f, 0.f, 0.f, 0.f, 0.f};
    c0 = WMMA_BF16(a, bb, c0);
#pragma unroll
    for (int r = 0; r < 8; ++r) {
      const int oc = mt2 * 16 + (lane >> 4) * 8 + r;
      const int px = pt * 128 + nt2 * 16 + (lane & 15);
      AGG[((size_t)b * 768 + g * 32 + oc) * 1024 + px] = c0[r];
    }
  }
}

// ---------- LiteMLA linear attention per (b, head): KV = relu(K)^T Vpad ; out = relu(Q) KV ----------
__global__ __launch_bounds__(256) void attn_kernel(
    const float* __restrict__ QKV, const float* __restrict__ AGG,
    float* __restrict__ OUT) {
  __shared__ float smem[32 * 128 + 48 * 128 + 32 * 48];  // sk | sv | skv  (47 KB)
  float* sk = smem;
  float* sv = smem + 32 * 128;
  float* skv = smem + 32 * 128 + 48 * 128;
  const int tid = threadIdx.x, wave = tid >> 5, lane = tid & 31;
  const int b = blockIdx.x >> 4, nh = blockIdx.x & 15;
  const float* src = (nh < 8) ? QKV : AGG;
  const int hb = (nh & 7) * 96;
  const float* qg = src + ((size_t)b * 768 + hb) * 1024;
  const float* kg = qg + 32 * 1024;
  const float* vg = qg + 64 * 1024;

  // ---- phase 1: KV[32 x 33(pad 48)] over 1024 pixels; waves 0..5 own one 16x16 tile ----
  v8f acc = {0.f, 0.f, 0.f, 0.f, 0.f, 0.f, 0.f, 0.f};
  const int mt = wave / 3, et = wave % 3;   // valid for wave < 6
  for (int p0 = 0; p0 < 1024; p0 += 128) {
    for (int i = tid; i < 32 * 128; i += 256) {
      const float kval = kg[(size_t)(i >> 7) * 1024 + p0 + (i & 127)];
      sk[i] = kval > 0.f ? kval : 0.f;
    }
    for (int i = tid; i < 48 * 128; i += 256) {
      const int e = i >> 7, pl = i & 127;
      sv[i] = (e < 32) ? vg[(size_t)e * 1024 + p0 + pl] : (e == 32 ? 1.0f : 0.0f);
    }
    __syncthreads();
    if (wave < 6) {
#pragma unroll
      for (int kk = 0; kk < 4; ++kk) {
        v16bf a = load_a_frag(sk + kk * 32, mt * 16, 128);    // A[d, p]
        v16bf bb = load_b_fragT(sv + kk * 32, et * 16, 128);  // B[p, e] (sv is [e][p])
        acc = WMMA_BF16(a, bb, acc);
      }
    }
    __syncthreads();
  }
  if (wave < 6) {
#pragma unroll
    for (int r = 0; r < 8; ++r) {
      const int d = mt * 16 + (lane >> 4) * 8 + r;
      const int e = et * 16 + (lane & 15);
      skv[d * 48 + e] = acc[r];
    }
  }
  __syncthreads();

  // ---- phase 2: out[p, e] = relu(Q)[p, d] * KV[d, e]; normalize by column 32 ----
  float* sout = smem + (size_t)wave * (16 * 48);  // reuse sk/sv region per wave
  for (int job = wave; job < 64; job += 8) {
    const int pb = job * 16;
    // A frag: row = pixel, col = d  (transposed gather from (ch, px) layout, with relu)
    v16bf af;
    {
      const int g = lane >> 4;
      const int p = pb + (lane & 15);
#pragma unroll
      for (int h = 0; h < 8; ++h) {
        const float q0 = qg[(size_t)(g * 16 + h) * 1024 + p];
        af[h] = (bf16_t)(q0 > 0.f ? q0 : 0.f);
        const float q1 = qg[(size_t)(g * 16 + 8 + h) * 1024 + p];
        af[h + 8] = (bf16_t)(q1 > 0.f ? q1 : 0.f);
      }
    }
#pragma unroll
    for (int e3 = 0; e3 < 3; ++e3) {
      v16bf bf_ = load_b_frag(skv, e3 * 16, 48);
      v8f c0 = {0.f, 0.f, 0.f, 0.f, 0.f, 0.f, 0.f, 0.f};
      c0 = WMMA_BF16(af, bf_, c0);
#pragma unroll
      for (int r = 0; r < 8; ++r) {
        const int m = (lane >> 4) * 8 + r;
        sout[m * 48 + e3 * 16 + (lane & 15)] = c0[r];
      }
    }
    for (int i = lane; i < 16 * 32; i += 32) {
      const int m = i >> 5, d = i & 31;
      const float den = sout[m * 48 + 32] + 1e-5f;
      OUT[((size_t)b * 512 + nh * 32 + d) * 1024 + pb + m] = sout[m * 48 + d] / den;
    }
  }
}

// ---------- MBConv depthwise 3x3 + bias + hardswish ----------
__global__ __launch_bounds__(256) void dw3_kernel(
    const float* __restrict__ IN, const float* __restrict__ Wd,
    const float* __restrict__ Bd, float* __restrict__ O) {
  const size_t idx = (size_t)blockIdx.x * 256 + threadIdx.x;  // B*1024*1024
  const int px = (int)(idx & 1023);
  const size_t bc = idx >> 10;
  const int c = (int)(bc & 1023);
  const int y = px >> 5, x = px & 31;
  const float* im = IN + bc * 1024;
  const float* w = Wd + c * 9;
  float s = Bd[c];
#pragma unroll
  for (int dy = -1; dy <= 1; ++dy) {
    const int yy = y + dy;
    if (yy < 0 || yy > 31) continue;
#pragma unroll
    for (int dx = -1; dx <= 1; ++dx) {
      const int xx = x + dx;
      if (xx < 0 || xx > 31) continue;
      s += im[yy * 32 + xx] * w[(dy + 1) * 3 + (dx + 1)];
    }
  }
  O[idx] = hswish(s);
}

// ---------- host launcher ----------
extern "C" void kernel_launch(void* const* d_in, const int* in_sizes, int n_in,
                              void* d_out, int out_size, void* d_ws, size_t ws_size,
                              hipStream_t stream) {
  (void)in_sizes; (void)n_in; (void)out_size; (void)ws_size;
  const float* x        = (const float*)d_in[0];
  const float* w_qkv    = (const float*)d_in[1];
  const float* w_agg_dw = (const float*)d_in[2];
  const float* w_agg_pw = (const float*)d_in[3];
  const float* w_proj   = (const float*)d_in[4];
  const float* bnp_s    = (const float*)d_in[5];
  const float* bnp_b    = (const float*)d_in[6];
  const float* w_mb1    = (const float*)d_in[7];
  const float* b_mb1    = (const float*)d_in[8];
  const float* w_mb_dw  = (const float*)d_in[9];
  const float* b_mb_dw  = (const float*)d_in[10];
  const float* w_mb2    = (const float*)d_in[11];
  const float* bn2_s    = (const float*)d_in[12];
  const float* bn2_b    = (const float*)d_in[13];
  float* out = (float*)d_out;

  char* ws = (char*)d_ws;
  float* qkv  = (float*)(ws + 0);                        // 48 MB : (16,768,1024)
  float* agg  = (float*)(ws + (size_t)50331648);         // 48 MB : (16,768,1024)
  float* attn = (float*)(ws + (size_t)100663296);        // 32 MB : (16,512,1024)
  float* y    = (float*)(ws + (size_t)134217728);        // 16 MB : (16,256,1024)
  float* h1   = (float*)(ws + 0);                        // 64 MB (reuses qkv/agg)
  float* h2   = (float*)(ws + (size_t)67108864);         // 64 MB (reuses agg/attn)

  const dim3 blk(256);
  // 1. qkv = Wqkv * x                 (M=768, K=256)
  gemm_wmma_kernel<0><<<dim3(16 * 8, 12), blk, 0, stream>>>(
      w_qkv, x, qkv, nullptr, nullptr, nullptr, 256, 768);
  // 2. agg = grouped1x1(dw5x5(qkv))
  agg_kernel<<<dim3(16 * 24, 8), blk, 0, stream>>>(qkv, w_agg_dw, w_agg_pw, agg);
  // 3. linear attention -> attn (16,512,1024)
  attn_kernel<<<dim3(16 * 16), blk, 0, stream>>>(qkv, agg, attn);
  // 4. y = x + BN(Wproj * attn)       (M=256, K=512)
  gemm_wmma_kernel<2><<<dim3(16 * 8, 4), blk, 0, stream>>>(
      w_proj, attn, y, bnp_s, bnp_b, x, 512, 256);
  // 5. h1 = hswish(Wmb1 * y + b)      (M=1024, K=256)
  gemm_wmma_kernel<1><<<dim3(16 * 8, 16), blk, 0, stream>>>(
      w_mb1, y, h1, b_mb1, nullptr, nullptr, 256, 1024);
  // 6. h2 = hswish(dw3x3(h1) + b)
  dw3_kernel<<<dim3(65536), blk, 0, stream>>>(h1, w_mb_dw, b_mb_dw, h2);
  // 7. out = y + BN(Wmb2 * h2)        (M=256, K=1024)
  gemm_wmma_kernel<2><<<dim3(16 * 8, 4), blk, 0, stream>>>(
      w_mb2, h2, out, bn2_s, bn2_b, y, 1024, 256);
}
